// TriangleSelfAttention_37134287242028
// MI455X (gfx1250) — matrix operand order repaired and checked by reading, not compiled, not run on hardware
//
#include <hip/hip_runtime.h>

#define N_NODES 768
#define KNN     32
#define CS      384
#define CZ      128
#define CG      16
#define NHEAD   4
#define DHEAD   32
#define NRBF    64
#define LN_EPS   1e-5f
#define DIST_EPS 1e-8f

// RBF constants: mu = linspace(0,20,64) -> step 20/63 ; sigma = 20/64
#define MU_STEP   (20.0f / 63.0f)
#define INV_SIGMA (64.0f / 20.0f)
#define INV_SQRT_DH 0.17677669529663687f   // 1/sqrt(32)

typedef __attribute__((ext_vector_type(16))) __bf16        v16bf;
typedef __attribute__((ext_vector_type(8)))  float         v8f;
typedef __attribute__((ext_vector_type(4)))  unsigned int  v4u;
typedef __attribute__((ext_vector_type(8)))  unsigned int  v8u;

// ---------------- small helpers ----------------

__device__ __forceinline__ unsigned short bfbits(float f) {
  return __builtin_bit_cast(unsigned short, (__bf16)f);   // native f32->bf16 cvt
}
__device__ __forceinline__ __bf16 u2bf(unsigned short s) {
  return __builtin_bit_cast(__bf16, s);
}
__device__ __forceinline__ float bf2f(unsigned short s) {
  union { float f; unsigned u; } v; v.u = ((unsigned)s) << 16; return v.f;
}
__device__ __forceinline__ float fast_rcp(float x) {
  return __builtin_amdgcn_rcpf(x);        // v_rcp_f32 (1 ULP, plenty for bf16 path)
}
__device__ __forceinline__ float sigmoidf(float x) {
  return fast_rcp(1.0f + __expf(-x));
}
__device__ __forceinline__ v8f zero8() {
  v8f z = {0.f, 0.f, 0.f, 0.f, 0.f, 0.f, 0.f, 0.f};
  return z;
}
__device__ __forceinline__ v8f wmma_bf16(v16bf a, v16bf b, v8f c) {
  return __builtin_amdgcn_wmma_f32_16x16x32_bf16(
      /*neg_a=*/false, a, /*neg_b=*/false, b,
      /*c_mod=*/(short)0, c, /*reuse_a=*/false, /*reuse_b=*/false);
}

// A fragment (16x32 bf16, MxK), buf row-major [m][k], row stride ld (ushorts).
// lane<16: m=row0+lane,     K = k0+{0..7, 16..23}
// lane>=16: m=row0+lane-16, K = k0+{8..15, 24..31}
__device__ __forceinline__ v16bf ldA(const unsigned short* buf, int ld,
                                     int row0, int k0, int lane) {
  int m  = row0 + (lane & 15);
  int kb = k0 + ((lane >> 4) << 3);
  const unsigned short* p = buf + m * ld + kb;
  v4u lo = *(const v4u*)p;          // K offs +0..7
  v4u hi = *(const v4u*)(p + 16);   // K offs +16..23
  v8u u;
#pragma unroll
  for (int t = 0; t < 4; ++t) { u[t] = lo[t]; u[4 + t] = hi[t]; }
  return __builtin_bit_cast(v16bf, u);
}

// B fragment from fragment-major packed buffer: packed[tile][lane][16] bf16.
__device__ __forceinline__ v16bf ldBpk(const unsigned short* buf, int tile, int lane) {
  v8u u = *(const v8u*)(buf + (((size_t)tile * 32 + lane) << 4));
  return __builtin_bit_cast(v16bf, u);
}

// B fragment where source is stored transposed: B[k][n] = buf[n*ld + k]
__device__ __forceinline__ v16bf ldBT(const unsigned short* buf, int ld,
                                      int k0, int col0, int lane) {
  int n  = col0 + (lane & 15);
  int kb = k0 + ((lane >> 4) << 3);
  const unsigned short* p = buf + n * ld + kb;
  v4u lo = *(const v4u*)p;
  v4u hi = *(const v4u*)(p + 16);
  v8u u;
#pragma unroll
  for (int t = 0; t < 4; ++t) { u[t] = lo[t]; u[4 + t] = hi[t]; }
  return __builtin_bit_cast(v16bf, u);
}

// B fragment scalar gather from row-major [k][n] LDS (cold path only).
__device__ __forceinline__ v16bf ldB_lds(const unsigned short* buf, int ld,
                                         int k0, int col0, int lane) {
  v16bf b;
  int n  = col0 + (lane & 15);
  int kb = k0 + ((lane >> 4) << 3);
#pragma unroll
  for (int t = 0; t < 16; ++t) b[t] = u2bf(buf[(kb + (t & 7) + ((t >> 3) << 4)) * ld + n]);
  return b;
}

// D fragment store (16x16 f32 -> bf16 LDS).
__device__ __forceinline__ void stD_bf16(unsigned short* buf, int ld,
                                         int row0, int col0, v8f d, int lane) {
  int n  = col0 + (lane & 15);
  int mb = row0 + ((lane >> 4) << 3);
#pragma unroll
  for (int r = 0; r < 8; ++r) buf[(mb + r) * ld + n] = bfbits(d[r]);
}

// ---------------- weight fp32 -> bf16 fragment-major packing ----------------
__global__ void pack_bfrag_kernel(const float* __restrict__ src,
                                  unsigned short* __restrict__ dst, int K, int N) {
  int o = blockIdx.x * blockDim.x + threadIdx.x;
  if (o >= K * N) return;
  int frag = o >> 9;
  int rem  = o & 511;
  int lane = rem >> 4;
  int t    = rem & 15;
  int ntiles = N >> 4;
  int kc = frag / ntiles;
  int nt = frag - kc * ntiles;
  int n = nt * 16 + (lane & 15);
  int k = kc * 32 + ((lane >> 4) << 3) + (t & 7) + ((t >> 3) << 4);
  dst[o] = bfbits(src[k * N + n]);
}

// ---------------- node left/right projections (tiny GEMMs) ----------------
__global__ void nodeproj_kernel(const float* __restrict__ nf,
                                const float* __restrict__ wl, const float* __restrict__ bl,
                                const float* __restrict__ wr, const float* __restrict__ br,
                                float* __restrict__ nl, float* __restrict__ nr) {
  int n = blockIdx.x;
  int t = threadIdx.x;      // 32 threads
  const float* w = (t < 16) ? wl : wr;
  const float* b = (t < 16) ? bl : br;
  int c = t & 15;
  float acc = b[c];
  const float* x = nf + n * CS;
  for (int k = 0; k < CS; ++k) acc += x[k] * w[k * CG + c];
  if (t < 16) nl[n * CG + c] = acc;
  else        nr[n * CG + c] = acc;
}

// ---------------- main fused per-node kernel ----------------
__global__ __launch_bounds__(256)
void tri_attn_kernel(const float* __restrict__ edge_features,
                     const long long* __restrict__ edge_index,
                     const float* __restrict__ node_trans,
                     const float* __restrict__ ln_g, const float* __restrict__ ln_b,
                     const float* __restrict__ nl, const float* __restrict__ nr,
                     const unsigned short* __restrict__ wbgp, const float* __restrict__ b_bgate,
                     const unsigned short* __restrict__ wdistp, const float* __restrict__ b_dist,
                     const float* __restrict__ w_tobias,
                     const unsigned short* __restrict__ wqp, const float* __restrict__ b_q,
                     const unsigned short* __restrict__ wkvp, const float* __restrict__ b_kv,
                     const unsigned short* __restrict__ wogp, const float* __restrict__ b_ogate,
                     const unsigned short* __restrict__ woutp, const float* __restrict__ b_out,
                     float* __restrict__ out) {
  __shared__ __align__(32) unsigned short zh[KNN * CZ];          // 8KB   LN(edge feats) bf16
  __shared__ float e1f[KNN][CG];                                 // 2KB
  __shared__ float e2f[KNN][CG];                                 // 2KB
  __shared__ float tpos[KNN][4];                                 // 0.5KB
  __shared__ float distm[KNN][KNN];                              // 4KB
  __shared__ __align__(32) unsigned short wdl[NRBF * CZ];        // 16KB  packed w_dist frags
  __shared__ float wtb[CZ][NHEAD];                               // 2KB
  __shared__ __align__(32) unsigned short sgb[8][16 * CZ];       // 32KB  sigmoid(gate) / wave
  __shared__ float biasb[KNN][KNN][NHEAD];                       // 16KB
  __shared__ __align__(32) unsigned short qh[KNN * CZ];          // 8KB
  __shared__ __align__(32) unsigned short kh[KNN * CZ];          // 8KB
  __shared__ __align__(32) unsigned short vh[KNN * CZ];          // 8KB
  __shared__ __align__(32) unsigned short ogb[KNN * CZ];         // 8KB   sigmoid(out-gate)
  __shared__ float sc[NHEAD][KNN][KNN];                          // 16KB
  __shared__ __align__(32) unsigned short pb[NHEAD][KNN * KNN];  // 8KB   softmax probs bf16
  __shared__ __align__(32) unsigned short ob[KNN * CZ];          // 8KB   gated attn out bf16
  __shared__ int knn[KNN];

  const int node = blockIdx.x;
  const int tid  = threadIdx.x;
  const int lane = tid & 31;
  const int wave = tid >> 5;

  // -------- setup: prefetch, gathers, LN, distances --------
  if (tid < KNN) {
    __builtin_prefetch(edge_features + (size_t)(node * KNN + tid) * CZ, 0, 1);
    int s = (int)edge_index[node * KNN + tid];     // src row of [2,E]
    knn[tid] = s;
    tpos[tid][0] = node_trans[s * 3 + 0];
    tpos[tid][1] = node_trans[s * 3 + 1];
    tpos[tid][2] = node_trans[s * 3 + 2];
  }
  __syncthreads();

  for (int r = tid; r < KNN * CG; r += 256) {
    int i = r >> 4, k = r & 15;
    int s = knn[i];
    e1f[i][k] = nl[s * CG + k];
    e2f[i][k] = nr[s * CG + k];
  }
  for (int r = tid; r < NRBF * CZ; r += 256) wdl[r] = wdistp[r];   // packed copy
  for (int r = tid; r < CZ * NHEAD; r += 256) (&wtb[0][0])[r] = w_tobias[r];

  // LayerNorm: wave w rows {w, w+8, w+16, w+24}; lane handles 4 contiguous cols
  for (int rr = 0; rr < 4; ++rr) {
    int i = wave + rr * 8;
    const float4* x4 = (const float4*)(edge_features + (size_t)(node * KNN + i) * CZ);
    float4 v4 = x4[lane];
    float s  = v4.x + v4.y + v4.z + v4.w;
    float s2 = v4.x * v4.x + v4.y * v4.y + v4.z * v4.z + v4.w * v4.w;
#pragma unroll
    for (int off = 16; off >= 1; off >>= 1) {
      s  += __shfl_xor(s,  off);
      s2 += __shfl_xor(s2, off);
    }
    float mu   = s  * (1.0f / CZ);
    float var  = s2 * (1.0f / CZ) - mu * mu;
    float rstd = rsqrtf(var + LN_EPS);
    const float4 g4 = ((const float4*)ln_g)[lane];
    const float4 b4 = ((const float4*)ln_b)[lane];
    unsigned short* zp = zh + i * CZ + lane * 4;
    zp[0] = bfbits((v4.x - mu) * rstd * g4.x + b4.x);
    zp[1] = bfbits((v4.y - mu) * rstd * g4.y + b4.y);
    zp[2] = bfbits((v4.z - mu) * rstd * g4.z + b4.z);
    zp[3] = bfbits((v4.w - mu) * rstd * g4.w + b4.w);
  }

  for (int r = tid; r < KNN * KNN; r += 256) {
    int i = r >> 5, j = r & 31;
    float dx = tpos[i][0] - tpos[j][0] + DIST_EPS;
    float dy = tpos[i][1] - tpos[j][1] + DIST_EPS;
    float dz = tpos[i][2] - tpos[j][2] + DIST_EPS;
    distm[i][j] = sqrtf(dx * dx + dy * dy + dz * dz);
  }
  __syncthreads();

  // -------- phase 1: bias[i,j,h] = (sigmoid(edge3@Wb + bb) * (rbf@Wd + bd)) @ Wt --------
  for (int s8 = 0; s8 < 8; ++s8) {
    int mt   = wave * 8 + s8;
    int mrow = mt * 16 + (lane & 15);
    int gi = mrow >> 5, gj = mrow & 31;
    int loff = (lane >> 4) << 3;

    float e2v[8];
#pragma unroll
    for (int t = 0; t < 8; ++t) e2v[t] = e2f[gj][loff + t];

    // gate = edge3 @ w_bgate : A built on the fly from e1 (x) e2, K=256
    v8f acc[8];
#pragma unroll
    for (int nt = 0; nt < 8; ++nt) acc[nt] = zero8();
#pragma unroll 1
    for (int kc = 0; kc < 8; ++kc) {
      float e1a = e1f[gi][2 * kc];
      float e1b = e1f[gi][2 * kc + 1];
      v16bf a;
#pragma unroll
      for (int t = 0; t < 8; ++t) {
        a[t]     = (__bf16)(e1a * e2v[t]);
        a[8 + t] = (__bf16)(e1b * e2v[t]);
      }
#pragma unroll
      for (int nt = 0; nt < 8; ++nt)
        acc[nt] = wmma_bf16(a, ldBpk(wbgp, kc * 8 + nt, lane), acc[nt]);
    }
    // sigmoid(gate + b_bgate) -> per-wave LDS staging
#pragma unroll
    for (int nt = 0; nt < 8; ++nt) {
      int n = nt * 16 + (lane & 15);
      float bb = b_bgate[n];
#pragma unroll
      for (int r = 0; r < 8; ++r) {
        int m = ((lane >> 4) << 3) + r;
        sgb[wave][m * CZ + n] = bfbits(sigmoidf(acc[nt][r] + bb));
      }
    }

    // db = rbf(dist) @ w_dist : A (rbf) built on the fly, K=64
#pragma unroll
    for (int nt = 0; nt < 8; ++nt) acc[nt] = zero8();
    float d = distm[gi][gj];
#pragma unroll 1
    for (int kc = 0; kc < 2; ++kc) {
      v16bf a;
      int kb = kc * 32 + loff;
#pragma unroll
      for (int t = 0; t < 8; ++t) {
        float x1 = (d - (float)(kb + t)      * MU_STEP) * INV_SIGMA;
        float x2 = (d - (float)(kb + 16 + t) * MU_STEP) * INV_SIGMA;
        a[t]     = (__bf16)__expf(-x1 * x1);
        a[8 + t] = (__bf16)__expf(-x2 * x2);
      }
#pragma unroll
      for (int nt = 0; nt < 8; ++nt)
        acc[nt] = wmma_bf16(a, ldBpk(wdl, kc * 8 + nt, lane), acc[nt]);
    }

    // (sig * db) @ w_tobias -> bias[i,j,h] (reduce c over half-wave)
    float bacc[8][4];
#pragma unroll
    for (int r = 0; r < 8; ++r)
#pragma unroll
      for (int h = 0; h < 4; ++h) bacc[r][h] = 0.f;
#pragma unroll
    for (int nt = 0; nt < 8; ++nt) {
      int n = nt * 16 + (lane & 15);
      float bd = b_dist[n];
#pragma unroll
      for (int r = 0; r < 8; ++r) {
        int m = ((lane >> 4) << 3) + r;
        float sg  = bf2f(sgb[wave][m * CZ + n]);
        float pdb = sg * (acc[nt][r] + bd);
#pragma unroll
        for (int h = 0; h < 4; ++h) bacc[r][h] += pdb * wtb[n][h];
      }
    }
#pragma unroll
    for (int r = 0; r < 8; ++r)
#pragma unroll
      for (int h = 0; h < 4; ++h) {
        float v = bacc[r][h];
        v += __shfl_xor(v, 1);
        v += __shfl_xor(v, 2);
        v += __shfl_xor(v, 4);
        v += __shfl_xor(v, 8);
        bacc[r][h] = v;
      }
    if ((lane & 15) == 0) {
      int mbase = mt * 16 + ((lane >> 4) << 3);
#pragma unroll
      for (int r = 0; r < 8; ++r) {
        int row = mbase + r;
        int i = row >> 5, j = row & 31;
#pragma unroll
        for (int h = 0; h < 4; ++h) biasb[i][j][h] = bacc[r][h];
      }
    }
  }

  // -------- phase 2: q / kv / out-gate projections (z @ W + b) --------
  for (int jj = 0; jj < 8; ++jj) {
    int job = wave * 8 + jj;
    const unsigned short* Bp;
    const float* bvec;
    int mt, bcol, kind, ntilesN;
    if (job < 16)      { kind = 0; mt = job >> 3;  bcol = (job & 7) * 16;  Bp = wqp;  bvec = b_q;     ntilesN = 8; }
    else if (job < 48) { int j2 = job - 16; kind = 1; mt = j2 >> 4; bcol = (j2 & 15) * 16; Bp = wkvp; bvec = b_kv; ntilesN = 16; }
    else               { int j2 = job - 48; kind = 2; mt = j2 >> 3; bcol = (j2 & 7) * 16;  Bp = wogp; bvec = b_ogate; ntilesN = 8; }
    int ntt = bcol >> 4;

    v8f acc = zero8();
#pragma unroll
    for (int kc = 0; kc < 4; ++kc) {
      v16bf a = ldA(zh, CZ, mt * 16, kc * 32, lane);
      v16bf b = ldBpk(Bp, kc * ntilesN + ntt, lane);
      acc = wmma_bf16(a, b, acc);
    }
    int n = bcol + (lane & 15);
    float bb = bvec[n];
    v8f accb;
    if (kind == 2) {
#pragma unroll
      for (int r = 0; r < 8; ++r) accb[r] = sigmoidf(acc[r] + bb);
      stD_bf16(ogb, CZ, mt * 16, bcol, accb, lane);
    } else {
#pragma unroll
      for (int r = 0; r < 8; ++r) accb[r] = acc[r] + bb;
      if (kind == 0)      stD_bf16(qh, CZ, mt * 16, bcol, accb, lane);
      else if (bcol < CZ) stD_bf16(kh, CZ, mt * 16, bcol, accb, lane);
      else                stD_bf16(vh, CZ, mt * 16, bcol - CZ, accb, lane);
    }
  }
  __syncthreads();

  // -------- phase 3: scores[h,i,j] = q.kT/sqrt(dh) + bias --------
  for (int jj = 0; jj < 2; ++jj) {
    int job = wave * 2 + jj;            // 16 jobs: h(4) x it(2) x jt(2)
    int h  = job >> 2;
    int it = (job >> 1) & 1;
    int jt = job & 1;
    v16bf a = ldA(qh, CZ, it * 16, h * DHEAD, lane);
    v16bf b = ldBT(kh, CZ, h * DHEAD, jt * 16, lane);
    v8f acc = wmma_bf16(a, b, zero8());
    int n  = jt * 16 + (lane & 15);
    int mb = it * 16 + ((lane >> 4) << 3);
#pragma unroll
    for (int r = 0; r < 8; ++r) {
      int i = mb + r;
      sc[h][i][n] = acc[r] * INV_SQRT_DH + biasb[i][n][h];
    }
  }
  __syncthreads();

  // softmax over j : 128 rows (h,i), one row per thread
  if (tid < 128) {
    int h = tid >> 5, i = tid & 31;
    float mx = -1e30f;
#pragma unroll
    for (int j = 0; j < KNN; ++j) mx = fmaxf(mx, sc[h][i][j]);
    float e[KNN];
    float sum = 0.f;
#pragma unroll
    for (int j = 0; j < KNN; ++j) { float v = __expf(sc[h][i][j] - mx); e[j] = v; sum += v; }
    float inv = fast_rcp(sum);
#pragma unroll
    for (int j = 0; j < KNN; ++j) pb[h][i * KNN + j] = bfbits(e[j] * inv);
  }
  __syncthreads();

  // -------- phase 4: o = p @ v, multiply by sigmoid out-gate --------
  for (int jj = 0; jj < 2; ++jj) {
    int job = wave * 2 + jj;            // 16 jobs: it(2) x h(4) x dt(2)
    int it = job >> 3;
    int h  = (job >> 1) & 3;
    int dt = job & 1;
    v16bf a = ldA(pb[h], KNN, it * 16, 0, lane);
    v16bf b = ldB_lds(vh, CZ, 0, h * DHEAD + dt * 16, lane);
    v8f acc = wmma_bf16(a, b, zero8());
    int c  = h * DHEAD + dt * 16 + (lane & 15);
    int mb = it * 16 + ((lane >> 4) << 3);
#pragma unroll
    for (int r = 0; r < 8; ++r) {
      float sg = bf2f(ogb[(mb + r) * CZ + c]);
      ob[(mb + r) * CZ + c] = bfbits(acc[r] * sg);
    }
  }
  __syncthreads();

  // -------- phase 5: out = o @ w_out + b_out (f32 to HBM) --------
  for (int jj = 0; jj < 2; ++jj) {
    int job = wave * 2 + jj;            // 16 jobs: mt(2) x nt(8)
    int mt = job >> 3;
    int nt = job & 7;
    v8f acc = zero8();
#pragma unroll
    for (int kc = 0; kc < 4; ++kc) {
      v16bf a = ldA(ob, CZ, mt * 16, kc * 32, lane);
      v16bf b = ldBpk(woutp, kc * 8 + nt, lane);
      acc = wmma_bf16(a, b, acc);
    }
    int n  = nt * 16 + (lane & 15);
    float bo = b_out[n];
    int mb = mt * 16 + ((lane >> 4) << 3);
#pragma unroll
    for (int r = 0; r < 8; ++r)
      out[(size_t)(node * KNN + mb + r) * CZ + n] = acc[r] + bo;
  }
}

// ---------------- host launcher ----------------
extern "C" void kernel_launch(void* const* d_in, const int* in_sizes, int n_in,
                              void* d_out, int out_size, void* d_ws, size_t ws_size,
                              hipStream_t stream) {
  const float*     node_features = (const float*)d_in[0];
  const float*     node_trans    = (const float*)d_in[1];
  const float*     edge_features = (const float*)d_in[2];
  const long long* edge_index    = (const long long*)d_in[3];
  const float* w_left   = (const float*)d_in[4];
  const float* b_left   = (const float*)d_in[5];
  const float* w_right  = (const float*)d_in[6];
  const float* b_right  = (const float*)d_in[7];
  const float* w_bgate  = (const float*)d_in[8];
  const float* b_bgate  = (const float*)d_in[9];
  const float* w_dist   = (const float*)d_in[10];
  const float* b_dist   = (const float*)d_in[11];
  const float* w_tobias = (const float*)d_in[12];
  const float* ln_g     = (const float*)d_in[13];
  const float* ln_b     = (const float*)d_in[14];
  const float* w_q      = (const float*)d_in[15];
  const float* b_q      = (const float*)d_in[16];
  const float* w_kv     = (const float*)d_in[17];
  const float* b_kv     = (const float*)d_in[18];
  const float* w_out    = (const float*)d_in[19];
  const float* b_out    = (const float*)d_in[20];
  const float* w_ogate  = (const float*)d_in[21];
  const float* b_ogate  = (const float*)d_in[22];

  // workspace layout (packed bf16 weights first, then floats)
  unsigned short* wsu   = (unsigned short*)d_ws;
  unsigned short* wbgp  = wsu + 0;        // 256*128
  unsigned short* wdistp= wsu + 32768;    //  64*128
  unsigned short* wqp   = wsu + 40960;    // 128*128
  unsigned short* wkvp  = wsu + 57344;    // 128*256
  unsigned short* wogp  = wsu + 90112;    // 128*128
  unsigned short* woutp = wsu + 106496;   // 128*128  (ends at 122880 ushorts)
  float* fws = (float*)((char*)d_ws + 245760);
  float* nl  = fws;                 // 768*16
  float* nr  = fws + N_NODES * CG;  // 768*16

  pack_bfrag_kernel<<<(256 * 128 + 255) / 256, 256, 0, stream>>>(w_bgate, wbgp,  256, 128);
  pack_bfrag_kernel<<<( 64 * 128 + 255) / 256, 256, 0, stream>>>(w_dist,  wdistp, 64, 128);
  pack_bfrag_kernel<<<(128 * 128 + 255) / 256, 256, 0, stream>>>(w_q,     wqp,   128, 128);
  pack_bfrag_kernel<<<(128 * 256 + 255) / 256, 256, 0, stream>>>(w_kv,    wkvp,  128, 256);
  pack_bfrag_kernel<<<(128 * 128 + 255) / 256, 256, 0, stream>>>(w_ogate, wogp,  128, 128);
  pack_bfrag_kernel<<<(128 * 128 + 255) / 256, 256, 0, stream>>>(w_out,   woutp, 128, 128);

  nodeproj_kernel<<<N_NODES, 32, 0, stream>>>(node_features, w_left, b_left,
                                              w_right, b_right, nl, nr);

  tri_attn_kernel<<<N_NODES, 256, 0, stream>>>(
      edge_features, edge_index, node_trans, ln_g, ln_b, nl, nr,
      wbgp, b_bgate, wdistp, b_dist, w_tobias,
      wqp, b_q, wkvp, b_kv, wogp, b_ogate, woutp, b_out,
      (float*)d_out);
}
